// CostVolumeDepthPredictor_53480932770416
// MI455X (gfx1250) — matrix operand order repaired and checked by reading, not compiled, not run on hardware
//
#include <hip/hip_runtime.h>
#include <hip/hip_bf16.h>

// ---------------------------------------------------------------------------
// CostVolumeDepthPredictor for MI455X (gfx1250, wave32, WMMA).
// Dense convs = implicit GEMM on v_wmma_f32_16x16x32_bf16, f32 accumulation.
// A fragments pre-packed in register order (1x 32B global load / wmma);
// B fragments staged through LDS as [pixel][channel] bf16 (2x ds_load_b128).
// Blocks own a 2D (TRx16) output tile so the 3x3 halo is amortized across
// 4 rows and all co-tiles: input read amplification ~1.7x instead of ~3.4x.
// ---------------------------------------------------------------------------

#define V 4
#define CCH 64
#define DD 64
#define LH 48
#define LW 48
#define NPIX (LH*LW)
#define FH 192
#define FW 192
#define FPIX (FH*FW)
#define HIDC 32
#define GN_G 8
#define TEMP_ 0.35f
#define BETA_ 0.2f
#define MAXCIN 128          // max LDS channel slots (dh1 concat = 128)

typedef __bf16 bf16;
typedef __attribute__((ext_vector_type(16))) __bf16 v16bf;
typedef __attribute__((ext_vector_type(8)))  __bf16 v8bf;
typedef __attribute__((ext_vector_type(8)))  float  v8f;

// setup scratch layout (floats)
#define S_NEAR 0
#define S_FAR  1
#define S_SPAN 2
#define S_IMIN 3
#define S_IMAX 4
#define S_INVB 8      // 64 inv-depth bins
#define S_KINV 80     // V*9
#define S_P    128    // V*3*9
#define S_Q    240    // V*3*3
#define S_TOTAL 512

// d_in indices: 5 arrays then params flattened in alphabetical pytree order
enum {
  IN_LOWRES=0, IN_FULLRES=1, IN_IMAGES=2, IN_INTR=3, IN_EXTR=4,
  P_CR1_B=5, P_CR1_W=6, P_CR2_B=7, P_CR2_W=8,
  P_CRDS_DW=9, P_CRDS_GNB=10, P_CRDS_GNW=11, P_CRDS_PWB=12, P_CRDS_PWW=13,
  P_DH1_B=14, P_DH1_W=15, P_DH2_B=16, P_DH2_W=17,
  P_DHDS_DW=18, P_DHDS_GNB=19, P_DHDS_GNW=20, P_DHDS_PWB=21, P_DHDS_PWW=22,
  P_FR1_B=23, P_FR1_W=24, P_FR2_B=25, P_FR2_W=26,
  P_FRDS_DW=27, P_FRDS_GNB=28, P_FRDS_GNW=29, P_FRDS_PWB=30, P_FRDS_PWW=31,
  P_PC1_B=32, P_PC1_W=33, P_PC2_B=34, P_PC2_W=35
};

__device__ __forceinline__ bf16 f2bf(float f) {
  unsigned u = __builtin_bit_cast(unsigned, f);
  unsigned r = u + 0x7FFFu + ((u >> 16) & 1u);   // round to nearest even
  unsigned short h = (unsigned short)(r >> 16);
  return __builtin_bit_cast(bf16, h);
}
__device__ __forceinline__ float gelu_f(float x) {
  return 0.5f * x * (1.0f + erff(x * 0.70710678118654752f));
}
__device__ void inv3(const float* m, float* o) {
  float a=m[0],b=m[1],c=m[2],d=m[3],e=m[4],f=m[5],g=m[6],h=m[7],i=m[8];
  float A = e*i - f*h, B = -(d*i - f*g), Cc = d*h - e*g;
  float det = a*A + b*B + c*Cc;
  float id = 1.0f / det;
  o[0]=A*id;            o[1]=-(b*i-c*h)*id;   o[2]=(b*f-c*e)*id;
  o[3]=B*id;            o[4]=(a*i-c*g)*id;    o[5]=-(a*f-c*d)*id;
  o[6]=Cc*id;           o[7]=-(a*h-b*g)*id;   o[8]=(a*e-b*d)*id;
}

// ---------------------------------------------------------------------------
// Tiny scalar kernel: scene bounds, w2c, inv bins, K^-1, per-(ref,src) P,q
// ---------------------------------------------------------------------------
__global__ void setup_k(const float* __restrict__ intr, const float* __restrict__ extr,
                        float* __restrict__ S) {
  if (threadIdx.x != 0 || blockIdx.x != 0) return;
  float Cn[V][3], Fw[V][3];
  for (int v=0; v<V; ++v) {
    for (int r=0;r<3;++r){ Cn[v][r]=extr[v*16+r*4+3]; Fw[v][r]=extr[v*16+r*4+2]; }
    float nn = sqrtf(Fw[v][0]*Fw[v][0]+Fw[v][1]*Fw[v][1]+Fw[v][2]*Fw[v][2]);
    nn = fmaxf(nn, 1e-6f);
    for (int r=0;r<3;++r) Fw[v][r] /= nn;
  }
  float Nm[9]={0,0,0,0,0,0,0,0,0}, rhs[3]={0,0,0};
  for (int v=0;v<V;++v)
    for (int r=0;r<3;++r) for (int c=0;c<3;++c) {
      float pr = ((r==c)?1.0f:0.0f) - Fw[v][r]*Fw[v][c];
      Nm[r*3+c] += pr; rhs[r] += pr*Cn[v][c];
    }
  Nm[0]+=1e-3f; Nm[4]+=1e-3f; Nm[8]+=1e-3f;
  float Ninv[9]; inv3(Nm, Ninv);
  float focus[3];
  for (int r=0;r<3;++r) focus[r]=Ninv[r*3]*rhs[0]+Ninv[r*3+1]*rhs[1]+Ninv[r*3+2]*rhs[2];
  float Rw[V][9], tw[V][3], depths[V];
  for (int v=0;v<V;++v) {
    for (int r=0;r<3;++r) for (int c=0;c<3;++c) Rw[v][r*3+c]=extr[v*16+c*4+r]; // R^T
    for (int r=0;r<3;++r)
      tw[v][r] = -(Rw[v][r*3]*Cn[v][0]+Rw[v][r*3+1]*Cn[v][1]+Rw[v][r*3+2]*Cn[v][2]);
    depths[v]=Rw[v][6]*focus[0]+Rw[v][7]*focus[1]+Rw[v][8]*focus[2]+tw[v][2];
  }
  float pd[12]; int k=0;
  for (int a=0;a<V;++a) for (int b=0;b<V;++b) { if (a==b) continue;
    float dx=Cn[a][0]-Cn[b][0], dy=Cn[a][1]-Cn[b][1], dz=Cn[a][2]-Cn[b][2];
    pd[k++]=sqrtf(dx*dx+dy*dy+dz*dz); }
  for (int a=1;a<12;++a){ float x=pd[a]; int b=a-1; while(b>=0&&pd[b]>x){pd[b+1]=pd[b];--b;} pd[b+1]=x; }
  float baseline = 0.5f*(pd[5]+pd[6]);
  float ds[V]; for (int a=0;a<V;++a) ds[a]=depths[a];
  for (int a=1;a<V;++a){ float x=ds[a]; int b=a-1; while(b>=0&&ds[b]>x){ds[b+1]=ds[b];--b;} ds[b+1]=x; }
  float nearv = (ds[0]+0.3f*(ds[1]-ds[0]))*0.8f;      // quantile 0.1, linear
  float farv  = (ds[2]+0.7f*(ds[3]-ds[2]))*1.2f;      // quantile 0.9
  nearv = fminf(fmaxf(nearv, 0.1f), 99.75f);
  farv  = fmaxf(farv, nearv + fmaxf(baseline, 0.5f));
  farv  = fminf(fmaxf(farv, nearv+0.25f), 100.0f);
  float ia = 1.0f/farv, ib = 1.0f/nearv;
  for (int d=0; d<DD; ++d) S[S_INVB+d] = ia + (ib-ia)*((float)d/(float)(DD-1));
  S[S_NEAR]=nearv; S[S_FAR]=farv; S[S_IMIN]=ia; S[S_IMAX]=ib; S[S_SPAN]=fmaxf(ib-ia,1e-6f);
  for (int i=0;i<V;++i) {
    float Ks[9];
    for (int c=0;c<3;++c){ Ks[c]=intr[i*9+c]*(float)LW; Ks[3+c]=intr[i*9+3+c]*(float)LH; Ks[6+c]=intr[i*9+6+c]; }
    inv3(Ks, &S[S_KINV+i*9]);
  }
  for (int i=0;i<V;++i) {
    float Ri[9], ti[3];
    for (int r=0;r<3;++r){ for (int c=0;c<3;++c) Ri[r*3+c]=extr[i*16+r*4+c]; ti[r]=Cn[i][r]; }
    int s=0;
    for (int j=0;j<V;++j) { if (j==i) continue;
      float Ks[9];
      for (int c=0;c<3;++c){ Ks[c]=intr[j*9+c]*(float)LW; Ks[3+c]=intr[j*9+3+c]*(float)LH; Ks[6+c]=intr[j*9+6+c]; }
      float M1[9];
      for (int r=0;r<3;++r) for (int c=0;c<3;++c)
        M1[r*3+c]=Rw[j][r*3]*Ri[c]+Rw[j][r*3+1]*Ri[3+c]+Rw[j][r*3+2]*Ri[6+c];
      float* Pp=&S[S_P+(i*3+s)*9];
      for (int r=0;r<3;++r) for (int c=0;c<3;++c)
        Pp[r*3+c]=Ks[r*3]*M1[c]+Ks[r*3+1]*M1[3+c]+Ks[r*3+2]*M1[6+c];
      float tv[3];
      for (int r=0;r<3;++r)
        tv[r]=Rw[j][r*3]*ti[0]+Rw[j][r*3+1]*ti[1]+Rw[j][r*3+2]*ti[2]+tw[j][r];
      float* Qp=&S[S_Q+(i*3+s)*3];
      for (int r=0;r<3;++r) Qp[r]=Ks[r*3]*tv[0]+Ks[r*3+1]*tv[1]+Ks[r*3+2]*tv[2];
      ++s;
    }
  }
}

// ---------------------------------------------------------------------------
// Pack conv weights (O,I,k,k) f32 -> bf16 A fragments in WMMA register order:
// dst[coTile][tap][chunk][lane][e], e = 0..15 (lane's v16bf A fragment).
// A(16x32): lanes 0-15 hold K in {0..7,16..23}; lanes 16-31 -> +8.
// ---------------------------------------------------------------------------
__global__ void pack_w_k(const float* __restrict__ src, bf16* __restrict__ dst,
                         int O, int I, int KK) {
  const int nCh = (I + 31) >> 5;
  const int nCoT = O >> 4;
  long tot = (long)nCoT * KK * nCh * 512;
  long i = (long)blockIdx.x*blockDim.x + threadIdx.x;
  if (i >= tot) return;
  const int e    = (int)(i & 15);
  const int lane = (int)((i >> 4) & 31);
  long rest = i >> 9;
  const int ch  = (int)(rest % nCh);
  const int t   = (int)((rest / nCh) % KK);
  const int coT = (int)(rest / ((long)nCh*KK));
  const int half = lane >> 4, l16 = lane & 15;
  const int ka = ((e < 8) ? e : e + 8) + (half ? 8 : 0);
  const int ci = (ch << 5) + ka;
  const int co = (coT << 4) + l16;
  float v = (ci < I) ? src[((long)co*I + ci)*KK + t] : 0.0f;
  dst[i] = f2bf(v);
}

// ---------------------------------------------------------------------------
// Implicit-GEMM conv (ks=1|3, pad=ks/2) via v_wmma_f32_16x16x32_bf16.
// Block = TRx16 output tile (H%TR==0, W%16==0 for all layers here);
// waves = (Cout/16 co-tiles) x (TR row-subtiles). Halo tile staged once into
// LDS as [pixel][channel] bf16; B fragment = 2x 16B ds loads; A fragment =
// one 32B global load of pre-packed weights. Epilogue fuses bias/GELU/resid.
// ---------------------------------------------------------------------------
__global__ void conv_wmma_k(const float* __restrict__ X, const bf16* __restrict__ Apack,
                            const float* __restrict__ bias, float* __restrict__ Y,
                            const float* __restrict__ resid,
                            int Cin, int Cout, int H, int Wi, int ks, int act, int TR) {
  const int HW = H*Wi;
  const int nCh  = (Cin + 31) >> 5;
  const int Crow = nCh << 5;               // LDS channel slots per pixel
  const int xt = Wi >> 4;                  // 16-wide x tiles per row
  const int tilesPerN = (H / TR) * xt;
  const int n  = blockIdx.x / tilesPerN;
  const int tb = blockIdx.x % tilesPerN;
  const int py0 = (tb / xt) * TR;
  const int px0 = (tb % xt) << 4;
  const int lane = threadIdx.x & 31;
  const int wave = threadIdx.x >> 5;
  const int coT  = wave / TR;              // co-tile
  const int rsub = wave % TR;              // row within tile
  const int half = lane >> 4, l16 = lane & 15;
  const int pad = (ks == 3) ? 1 : 0;
  const int tcols = 16 + 2*pad;            // 18 or 16
  const int trows = TR + 2*pad;
  const int npx = tcols * trows;

  __shared__ __align__(32) bf16 sh[6*18*MAXCIN];   // 27 KB max (TR=4, ks=3)

  // stage halo tile: [pix][chan] bf16, zero-filled out of bounds / past Cin
  const float* Xn = X + (long)n*Cin*HW;
  for (int idx = threadIdx.x; idx < npx*Crow; idx += blockDim.x) {
    const int pp = idx % npx;
    const int c  = idx / npx;
    const int gy = py0 + pp / tcols - pad;
    const int gx = px0 + pp % tcols - pad;
    float v = 0.0f;
    if (c < Cin && (unsigned)gy < (unsigned)H && (unsigned)gx < (unsigned)Wi)
      v = Xn[(long)c*HW + gy*Wi + gx];
    sh[pp*Crow + c] = f2bf(v);
  }
  __syncthreads();

  const int taps = ks*ks;
  v8f acc = {};
  for (int t = 0; t < taps; ++t) {
    const int dy = (ks==3) ? (t/3 - 1) : 0;
    const int dx = (ks==3) ? (t%3 - 1) : 0;
    const int pix = (rsub + dy + pad)*tcols + (l16 + dx + pad);
    for (int ch = 0; ch < nCh; ++ch) {
      const v16bf a = *(const v16bf*)(Apack + ((((long)coT*taps + t)*nCh + ch)*32 + lane)*16);
      const int off = pix*Crow + (ch << 5) + (half << 3);
      const v8bf blo = *(const v8bf*)&sh[off];
      const v8bf bhi = *(const v8bf*)&sh[off + 16];
      v16bf b;
      #pragma unroll
      for (int e = 0; e < 8; ++e) { b[e] = blo[e]; b[e+8] = bhi[e]; }
      acc = __builtin_amdgcn_wmma_f32_16x16x32_bf16(false, a, false, b,
                                                    (short)0, acc, false, false);
    }
  }

  const int pcol = (py0 + rsub)*Wi + px0 + l16;
  #pragma unroll
  for (int r = 0; r < 8; ++r) {
    const int co = (coT << 4) + r + (half ? 8 : 0);  // C/D layout: M = r + 8*half
    float y = acc[r] + bias[co];
    if (act == 1) y = gelu_f(y);
    const long oidx = ((long)n*Cout + co)*HW + pcol;
    if (resid) y += resid[oidx];
    Y[oidx] = y;
  }
}

// ---------------------------------------------------------------------------
// VALU kernels
// ---------------------------------------------------------------------------
__global__ void norm_feats_k(const float* __restrict__ X, float* __restrict__ Y) {
  long i = (long)blockIdx.x*blockDim.x + threadIdx.x;
  if (i >= (long)V*NPIX) return;
  int v = (int)(i / NPIX), p = (int)(i % NPIX);
  const float* b = X + (long)v*CCH*NPIX + p;
  float s = 0.f;
  for (int c=0;c<CCH;++c){ float x=b[(long)c*NPIX]; s += x*x; }
  float inv = 1.0f / fmaxf(sqrtf(s), 1e-12f);
  float* o = Y + (long)v*CCH*NPIX + p;
  for (int c=0;c<CCH;++c) o[(long)c*NPIX] = b[(long)c*NPIX]*inv;
}

__global__ void cost_volume_k(const float* __restrict__ FN, const float* __restrict__ S,
                              float* __restrict__ AGG) {
  // block: 4 pixels x 64 depths; grid: V * (NPIX/4)
  const int i  = blockIdx.x / (NPIX/4);
  const int pg = blockIdx.x % (NPIX/4);
  const int d  = threadIdx.x & 63;
  const int lp = threadIdx.x >> 6;
  const int pix = pg*4 + lp;
  __shared__ float refS[4][CCH];
  { int c = threadIdx.x & 63; int l2 = threadIdx.x >> 6;
    refS[l2][c] = FN[((long)i*CCH + c)*NPIX + pg*4 + l2]; }
  __syncthreads();
  const int px = pix % LW, py = pix / LW;
  const float* Ki = S + S_KINV + i*9;
  const float rx = Ki[0]*px + Ki[1]*py + Ki[2];
  const float ry = Ki[3]*px + Ki[4]*py + Ki[5];
  const float rz = Ki[6]*px + Ki[7]*py + Ki[8];
  const float depth = 1.0f / S[S_INVB + d];
  float cost[3], vis[3];
  int s = 0;
  for (int j = 0; j < V; ++j) {
    if (j == i) continue;
    const float* P = S + S_P + (i*3+s)*9;
    const float* q = S + S_Q + (i*3+s)*3;
    float Xc = depth*(P[0]*rx+P[1]*ry+P[2]*rz) + q[0];
    float Yc = depth*(P[3]*rx+P[4]*ry+P[5]*rz) + q[1];
    float Zc = depth*(P[6]*rx+P[7]*ry+P[8]*rz) + q[2];
    float zs = fmaxf(Zc, 1e-6f);
    float u = Xc/zs, vv = Yc/zs;
    float vf = ((Zc > 1e-6f) && u >= 0.f && u <= (float)(LW-1) &&
                vv >= 0.f && vv <= (float)(LH-1)) ? 1.0f : 0.0f;
    float x0f = floorf(u), y0f = floorf(vv);
    float wx = u - x0f, wy = vv - y0f;
    float cw[4]; int cx[4], cy[4];
    #pragma unroll
    for (int t=0;t<4;++t) {
      float xf = x0f + (float)(t & 1);
      float yf = y0f + (float)(t >> 1);
      float inb = (xf >= 0.f && xf <= (float)(LW-1) && yf >= 0.f && yf <= (float)(LH-1)) ? 1.f : 0.f;
      float wxx = (t & 1) ? wx : (1.f - wx);
      float wyy = (t >> 1) ? wy : (1.f - wy);
      cw[t] = wxx * wyy * inb;
      cx[t] = (int)fminf(fmaxf(xf, 0.f), (float)(LW-1));
      cy[t] = (int)fminf(fmaxf(yf, 0.f), (float)(LH-1));
    }
    const float* Fj = FN + (long)j*CCH*NPIX;
    float acc = 0.f;
    for (int c=0;c<CCH;++c) {
      const float* Fc = Fj + (long)c*NPIX;
      float wc = cw[0]*Fc[cy[0]*LW+cx[0]] + cw[1]*Fc[cy[1]*LW+cx[1]]
               + cw[2]*Fc[cy[2]*LW+cx[2]] + cw[3]*Fc[cy[3]*LW+cx[3]];
      acc += wc * refS[lp][c];
    }
    float cs = 1.0f - acc;
    cost[s] = cs*vf + (1.0f - vf);
    vis[s] = vf;
    ++s;
  }
  float a0=-cost[0]/TEMP_, a1=-cost[1]/TEMP_, a2=-cost[2]/TEMP_;
  float mx = fmaxf(a0, fmaxf(a1, a2));
  float e0=expf(a0-mx), e1=expf(a1-mx), e2=expf(a2-mx);
  float Zs = e0+e1+e2;
  float w0=e0/Zs*(vis[0]+BETA_), w1=e1/Zs*(vis[1]+BETA_), w2=e2/Zs*(vis[2]+BETA_);
  float tot = w0+w1+w2+1e-6f;
  AGG[((long)i*DD + d)*NPIX + pix] = (w0*cost[0]+w1*cost[1]+w2*cost[2]) / tot;
}

__global__ void dwconv3_k(const float* __restrict__ X, const float* __restrict__ Wd,
                          float* __restrict__ Y, long total, int Cn, int H, int Wi) {
  long i = (long)blockIdx.x*blockDim.x + threadIdx.x;
  if (i >= total) return;
  const int HW = H*Wi;
  const int p = (int)(i % HW);
  const int c = (int)((i / HW) % Cn);
  const float* Xp = X + (i - p);
  const int py = p / Wi, px = p % Wi;
  float s = 0.f;
  #pragma unroll
  for (int t=0;t<9;++t) {
    int iy = py + t/3 - 1, ix = px + t%3 - 1;
    if ((unsigned)iy < (unsigned)H && (unsigned)ix < (unsigned)Wi)
      s += Wd[c*9+t] * Xp[iy*Wi+ix];
  }
  Y[i] = s;
}

__global__ void gn_stats_k(const float* __restrict__ X, float* __restrict__ stats,
                           int Cn, int HW) {
  const int n = blockIdx.x / GN_G, g = blockIdx.x % GN_G;
  const int cg = Cn / GN_G;
  const long cnt = (long)cg * HW;
  const float* base = X + ((long)n*Cn + (long)g*cg)*HW;
  float s=0.f, s2=0.f;
  for (long e = threadIdx.x; e < cnt; e += blockDim.x) { float x=base[e]; s+=x; s2+=x*x; }
  __shared__ float sh[256], sh2[256];
  sh[threadIdx.x]=s; sh2[threadIdx.x]=s2; __syncthreads();
  for (int o = 128; o > 0; o >>= 1) {
    if ((int)threadIdx.x < o) { sh[threadIdx.x]+=sh[threadIdx.x+o]; sh2[threadIdx.x]+=sh2[threadIdx.x+o]; }
    __syncthreads();
  }
  if (threadIdx.x == 0) {
    float m = sh[0] / (float)cnt;
    stats[blockIdx.x*2]   = m;
    stats[blockIdx.x*2+1] = sh2[0] / (float)cnt - m*m;
  }
}

__global__ void gn_apply_k(const float* __restrict__ X, const float* __restrict__ stats,
                           const float* __restrict__ gam, const float* __restrict__ bet,
                           const float* __restrict__ resid, float* __restrict__ Y,
                           long total, int Cn, int HW) {
  long i = (long)blockIdx.x*blockDim.x + threadIdx.x;
  if (i >= total) return;
  const int c = (int)((i / HW) % Cn);
  const int n = (int)(i / ((long)Cn*HW));
  const int g = c / (Cn / GN_G);
  const float m = stats[(n*GN_G+g)*2], v = stats[(n*GN_G+g)*2+1];
  float x = (X[i]-m) * rsqrtf(v + 1e-5f) * gam[c] + bet[c];
  Y[i] = gelu_f(x) + resid[i];
}

__global__ void concat_low_k(const float* __restrict__ A, const float* __restrict__ B,
                             float* __restrict__ Y) {
  long i = (long)blockIdx.x*blockDim.x + threadIdx.x;
  if (i >= (long)V*128*NPIX) return;
  const int p = (int)(i % NPIX);
  const int ch = (int)((i / NPIX) % 128);
  const int n = (int)(i / ((long)128*NPIX));
  Y[i] = (ch < 64) ? A[((long)n*64+ch)*NPIX+p] : B[((long)n*64+(ch-64))*NPIX+p];
}

__global__ void softmax_head_k(const float* __restrict__ L, const float* __restrict__ S,
                               float* __restrict__ inv_low, float* __restrict__ conf_low) {
  long i = (long)blockIdx.x*blockDim.x + threadIdx.x;
  if (i >= (long)V*NPIX) return;
  const int n = (int)(i / NPIX), p = (int)(i % NPIX);
  const float* b = L + (long)n*DD*NPIX + p;
  float mx = -3.4e38f;
  for (int d=0; d<DD; ++d) mx = fmaxf(mx, b[(long)d*NPIX]);
  float Z = 0.f, acc = 0.f;
  for (int d=0; d<DD; ++d) { float e = expf(b[(long)d*NPIX]-mx); Z += e; acc += e*S[S_INVB+d]; }
  inv_low[i]  = acc / Z;
  conf_low[i] = 1.0f / Z;   // max prob = exp(0)/Z
}

__global__ void head1_k(const float* __restrict__ X, const float* __restrict__ Wv,
                        const float* __restrict__ bp, float* __restrict__ Y,
                        int Cn, int HW, int act, long total) {
  long i = (long)blockIdx.x*blockDim.x + threadIdx.x;
  if (i >= total) return;
  const int n = (int)(i / HW), p = (int)(i % HW);
  float s = bp[0];
  for (int c=0;c<Cn;++c) s += Wv[c] * X[((long)n*Cn+c)*HW+p];
  if (act == 1) s = 1.0f / (1.0f + expf(-s));
  Y[i] = s;
}

__global__ void resize_k(const float* __restrict__ X, float* __restrict__ Y,
                         int Cn, int H, int Wi, int OH, int OW, int clip01, long total) {
  long i = (long)blockIdx.x*blockDim.x + threadIdx.x;
  if (i >= total) return;
  const int ox = (int)(i % OW);
  const int oy = (int)((i / OW) % OH);
  const int c  = (int)((i / ((long)OW*OH)) % Cn);
  const int n  = (int)(i / ((long)Cn*OH*OW));
  float sy = (float)oy * (float)(H-1) / (float)(OH-1);
  float sx = (float)ox * (float)(Wi-1) / (float)(OW-1);
  int y0 = (int)floorf(sy); y0 = y0 < 0 ? 0 : (y0 > H-2 ? H-2 : y0);
  int x0 = (int)floorf(sx); x0 = x0 < 0 ? 0 : (x0 > Wi-2 ? Wi-2 : x0);
  float wy = sy - (float)y0, wx = sx - (float)x0;
  const float* b = X + ((long)n*Cn + c)*H*Wi;
  float v00=b[y0*Wi+x0], v01=b[y0*Wi+x0+1], v10=b[(y0+1)*Wi+x0], v11=b[(y0+1)*Wi+x0+1];
  float val = v00*(1.f-wy)*(1.f-wx) + v01*(1.f-wy)*wx + v10*wy*(1.f-wx) + v11*wy*wx;
  if (clip01) val = fminf(fmaxf(val, 0.f), 1.f);
  Y[i] = val;
}

__global__ void concat_full_k(const float* __restrict__ FF, const float* __restrict__ IMG,
                              const float* __restrict__ INVF, float* __restrict__ Y) {
  long i = (long)blockIdx.x*blockDim.x + threadIdx.x;
  if (i >= (long)V*68*FPIX) return;
  const int p = (int)(i % FPIX);
  const int ch = (int)((i / FPIX) % 68);
  const int n = (int)(i / ((long)68*FPIX));
  float v;
  if (ch < 64)      v = FF[((long)n*64+ch)*FPIX+p];
  else if (ch < 67) v = IMG[((long)n*3+(ch-64))*FPIX+p];
  else              v = INVF[(long)n*FPIX+p];
  Y[i] = v;
}

__global__ void final_depth_k(const float* __restrict__ Yf, const float* __restrict__ w,
                              const float* __restrict__ b, const float* __restrict__ inv_full,
                              const float* __restrict__ S, float* __restrict__ out) {
  long i = (long)blockIdx.x*blockDim.x + threadIdx.x;
  if (i >= (long)V*FPIX) return;
  const int n = (int)(i / FPIX), p = (int)(i % FPIX);
  float s = b[0];
  for (int c=0;c<CCH;++c) s += w[c] * Yf[((long)n*CCH+c)*FPIX+p];
  float resid = 0.25f * S[S_SPAN] * tanhf(s);
  float iv = inv_full[i] + resid;
  iv = fminf(fmaxf(iv, S[S_IMIN]), S[S_IMAX]);
  out[i] = fmaxf(1.0f/iv, S[S_NEAR]);
}

// ---------------------------------------------------------------------------
// Host launch
// ---------------------------------------------------------------------------
static inline unsigned cdiv(long a, long b) { return (unsigned)((a + b - 1) / b); }
static inline size_t packSz(int O, int I, int KK) {
  return (size_t)(O >> 4) * KK * (size_t)((I + 31) >> 5) * 512;
}

extern "C" void kernel_launch(void* const* d_in, const int* in_sizes, int n_in,
                              void* d_out, int out_size, void* d_ws, size_t ws_size,
                              hipStream_t stream) {
  (void)in_sizes; (void)n_in; (void)out_size; (void)ws_size;
  const float* lowres  = (const float*)d_in[IN_LOWRES];
  const float* fullres = (const float*)d_in[IN_FULLRES];
  const float* images  = (const float*)d_in[IN_IMAGES];
  const float* intr    = (const float*)d_in[IN_INTR];
  const float* extr    = (const float*)d_in[IN_EXTR];

  float* ws = (float*)d_ws;
  size_t o = 0;
  float* Sp      = ws + o; o += S_TOTAL;
  float* gstat   = ws + o; o += 128;
  float* feats   = ws + o; o += (size_t)V*CCH*NPIX;
  float* agg     = ws + o; o += (size_t)V*DD*NPIX;
  float* t1      = ws + o; o += (size_t)V*CCH*NPIX;
  float* t2      = ws + o; o += (size_t)V*CCH*NPIX;
  float* t3      = ws + o; o += (size_t)V*CCH*NPIX;
  float* t4      = ws + o; o += (size_t)V*CCH*NPIX;
  float* t5      = ws + o; o += (size_t)V*CCH*NPIX;   // refined
  float* cin128  = ws + o; o += (size_t)V*128*NPIX;
  float* pchid   = ws + o; o += (size_t)V*HIDC*NPIX;
  float* inv_low = ws + o; o += (size_t)V*NPIX;
  float* conf_low= ws + o; o += (size_t)V*NPIX;
  float* cert_low= ws + o; o += (size_t)V*NPIX;
  float* inv_full= ws + o; o += (size_t)V*FPIX;
  float* concatIn= ws + o; o += (size_t)V*68*FPIX;    // reused as pw output fC
  float* fA      = ws + o; o += (size_t)V*CCH*FPIX;
  float* fB      = ws + o; o += (size_t)V*CCH*FPIX;
  bf16*  wreg    = (bf16*)(ws + o);
  size_t wo = 0;
  bf16* cr1b  = wreg + wo; wo += packSz(64,  64, 9);
  bf16* crpwb = wreg + wo; wo += packSz(64,  64, 1);
  bf16* cr2b  = wreg + wo; wo += packSz(64,  64, 1);
  bf16* dh1b  = wreg + wo; wo += packSz(64, 128, 9);
  bf16* dh2b  = wreg + wo; wo += packSz(64,  64, 1);
  bf16* dhpwb = wreg + wo; wo += packSz(64,  64, 1);
  bf16* pc1b  = wreg + wo; wo += packSz(32,  64, 9);
  bf16* fr1b  = wreg + wo; wo += packSz(64,  68, 9);
  bf16* frpwb = wreg + wo; wo += packSz(64,  64, 1);

  float* out_depth = (float*)d_out;
  float* out_cert  = out_depth + (size_t)V*FPIX;
  float* out_conf  = out_cert  + (size_t)V*FPIX;

  auto pack = [&](const void* src, bf16* dst, int O, int I, int KK) {
    long tot = (long)packSz(O, I, KK);
    pack_w_k<<<cdiv(tot,256), 256, 0, stream>>>((const float*)src, dst, O, I, KK);
  };
  auto conv = [&](const float* X, const bf16* Ap, const void* bias, float* Y,
                  const float* res, int N, int Cin, int Cout, int H, int Wd,
                  int ks, int act) {
    const int TR = (ks == 3) ? 4 : 2;      // rows per block tile
    int blocks = N * (H/TR) * (Wd >> 4);
    int thr = 32 * (Cout >> 4) * TR;       // one wave per (co-tile, row)
    conv_wmma_k<<<blocks, thr, 0, stream>>>(X, Ap, (const float*)bias, Y, res,
                                            Cin, Cout, H, Wd, ks, act, TR);
  };

  // ---- setup + weight packing ----
  setup_k<<<1, 1, 0, stream>>>(intr, extr, Sp);
  pack(d_in[P_CR1_W],   cr1b,  64,  64, 9);
  pack(d_in[P_CRDS_PWW],crpwb, 64,  64, 1);
  pack(d_in[P_CR2_W],   cr2b,  64,  64, 1);
  pack(d_in[P_DH1_W],   dh1b,  64, 128, 9);
  pack(d_in[P_DH2_W],   dh2b,  64,  64, 1);
  pack(d_in[P_DHDS_PWW],dhpwb, 64,  64, 1);
  pack(d_in[P_PC1_W],   pc1b,  32,  64, 9);
  pack(d_in[P_FR1_W],   fr1b,  64,  68, 9);
  pack(d_in[P_FRDS_PWW],frpwb, 64,  64, 1);

  // ---- cost volume ----
  norm_feats_k<<<cdiv((long)V*NPIX,256), 256, 0, stream>>>(lowres, feats);
  cost_volume_k<<<V*(NPIX/4), 256, 0, stream>>>(feats, Sp, agg);

  const long lowTot = (long)V*CCH*NPIX;

  // ---- cost refinement: cr1 -> ds_block -> cr2 (+agg residual) ----
  conv(agg, cr1b, d_in[P_CR1_B], t1, nullptr, V, DD, DD, LH, LW, 3, 1);
  dwconv3_k<<<cdiv(lowTot,256), 256, 0, stream>>>(t1, (const float*)d_in[P_CRDS_DW], t2, lowTot, DD, LH, LW);
  conv(t2, crpwb, d_in[P_CRDS_PWB], t3, nullptr, V, DD, DD, LH, LW, 1, 0);
  gn_stats_k<<<V*GN_G, 256, 0, stream>>>(t3, gstat, DD, NPIX);
  gn_apply_k<<<cdiv(lowTot,256), 256, 0, stream>>>(t3, gstat, (const float*)d_in[P_CRDS_GNW],
      (const float*)d_in[P_CRDS_GNB], t1, t4, lowTot, DD, NPIX);
  conv(t4, cr2b, d_in[P_CR2_B], t5, agg, V, DD, DD, LH, LW, 1, 0);   // refined

  // ---- depth head: concat -> dh1 -> ds_block -> dh2 -> softmax ----
  concat_low_k<<<cdiv((long)V*128*NPIX,256), 256, 0, stream>>>(lowres, t5, cin128);
  conv(cin128, dh1b, d_in[P_DH1_B], t1, nullptr, V, 128, CCH, LH, LW, 3, 1);
  dwconv3_k<<<cdiv(lowTot,256), 256, 0, stream>>>(t1, (const float*)d_in[P_DHDS_DW], t2, lowTot, CCH, LH, LW);
  conv(t2, dhpwb, d_in[P_DHDS_PWB], t3, nullptr, V, CCH, CCH, LH, LW, 1, 0);
  gn_stats_k<<<V*GN_G, 256, 0, stream>>>(t3, gstat, CCH, NPIX);
  gn_apply_k<<<cdiv(lowTot,256), 256, 0, stream>>>(t3, gstat, (const float*)d_in[P_DHDS_GNW],
      (const float*)d_in[P_DHDS_GNB], t1, t4, lowTot, CCH, NPIX);
  conv(t4, dh2b, d_in[P_DH2_B], t2, nullptr, V, CCH, DD, LH, LW, 1, 0); // logits -> t2
  softmax_head_k<<<cdiv((long)V*NPIX,256), 256, 0, stream>>>(t2, Sp, inv_low, conf_low);

  // ---- certainty head ----
  conv(t5, pc1b, d_in[P_PC1_B], pchid, nullptr, V, DD, HIDC, LH, LW, 3, 1);
  head1_k<<<cdiv((long)V*NPIX,256), 256, 0, stream>>>(pchid, (const float*)d_in[P_PC2_W],
      (const float*)d_in[P_PC2_B], cert_low, HIDC, NPIX, 1, (long)V*NPIX);

  // ---- full-res refinement ----
  resize_k<<<cdiv((long)V*FPIX,256), 256, 0, stream>>>(inv_low, inv_full, 1, LH, LW, FH, FW, 0, (long)V*FPIX);
  concat_full_k<<<cdiv((long)V*68*FPIX,256), 256, 0, stream>>>(fullres, images, inv_full, concatIn);
  conv(concatIn, fr1b, d_in[P_FR1_B], fA, nullptr, V, 68, CCH, FH, FW, 3, 1);
  const long fullTot = (long)V*CCH*FPIX;
  dwconv3_k<<<cdiv(fullTot,256), 256, 0, stream>>>(fA, (const float*)d_in[P_FRDS_DW], fB, fullTot, CCH, FH, FW);
  conv(fB, frpwb, d_in[P_FRDS_PWB], concatIn, nullptr, V, CCH, CCH, FH, FW, 1, 0); // fC = concatIn
  gn_stats_k<<<V*GN_G, 256, 0, stream>>>(concatIn, gstat, CCH, FPIX);
  gn_apply_k<<<cdiv(fullTot,256), 256, 0, stream>>>(concatIn, gstat, (const float*)d_in[P_FRDS_GNW],
      (const float*)d_in[P_FRDS_GNB], fA, fB, fullTot, CCH, FPIX);

  // ---- outputs ----
  final_depth_k<<<cdiv((long)V*FPIX,256), 256, 0, stream>>>(fB, (const float*)d_in[P_FR2_W],
      (const float*)d_in[P_FR2_B], inv_full, Sp, out_depth);
  resize_k<<<cdiv((long)V*FPIX,256), 256, 0, stream>>>(cert_low, out_cert, 1, LH, LW, FH, FW, 1, (long)V*FPIX);
  resize_k<<<cdiv((long)V*FPIX,256), 256, 0, stream>>>(conf_low, out_conf, 1, LH, LW, FH, FW, 1, (long)V*FPIX);
}
